// FixedDGCNNBackbone_59493886984495
// MI455X (gfx1250) — matrix operand order repaired and checked by reading, not compiled, not run on hardware
//
#include <hip/hip_runtime.h>
#include <hip/hip_bf16.h>

typedef __attribute__((ext_vector_type(16))) _Float16 v16h;
typedef __attribute__((ext_vector_type(8)))  float    v8f;
typedef __attribute__((ext_vector_type(4)))  unsigned int u32x4;
typedef __attribute__((ext_vector_type(8)))  int      i32x8;
typedef __attribute__((ext_vector_type(4)))  int      i32x4;

#define SLOPE 0.2f
#define BN_EPS 1e-5f
#define NK 20

// ---------------------------------------------------------------------------
// TDM: 2D f16 tile (tile_h rows x tile_w elems, row stride in elems) -> LDS.
// D# group0/group1 packed per CDNA5 ISA 8.3/8.4; groups 2/3/4 zero (2D tensor).
// This toolchain's builtin form (6 args): (u32x4, i32x8, i32x4, i32x4, i32x8, i32 cpol).
// ---------------------------------------------------------------------------
__device__ __forceinline__ void tdm_load_tile_f16(unsigned lds_off,
                                                  const _Float16* gbase,
                                                  unsigned tile_w, unsigned tile_h,
                                                  unsigned long long stride_elems) {
  unsigned long long ga = (unsigned long long)(uintptr_t)gbase;
  u32x4 g0;
  g0[0] = 1u;                                                 // count=1, user mode
  g0[1] = lds_off;                                            // lds_addr
  g0[2] = (unsigned)ga;                                       // global_addr[31:0]
  g0[3] = (unsigned)((ga >> 32) & 0x01FFFFFFu) | (2u << 30);  // addr[56:32] | type=2
  i32x8 g1;
  g1[0] = (int)(1u << 16);                                    // data_size = 2 bytes
  g1[1] = (int)((tile_w & 0xFFFFu) << 16);                    // tensor_dim0 lo
  g1[2] = (int)((tile_w >> 16) | ((tile_h & 0xFFFFu) << 16)); // dim0 hi | dim1 lo
  g1[3] = (int)((tile_h >> 16) | ((tile_w & 0xFFFFu) << 16)); // dim1 hi | tile_dim0
  g1[4] = (int)(tile_h & 0xFFFFu);                            // tile_dim1
  g1[5] = (int)(unsigned)stride_elems;                        // tensor_dim0_stride lo
  g1[6] = (int)((unsigned)(stride_elems >> 32) & 0xFFFFu);    // stride hi
  g1[7] = 0;
  i32x4 zz = {0, 0, 0, 0};
  i32x8 z8 = {0, 0, 0, 0, 0, 0, 0, 0};
  __builtin_amdgcn_tensor_load_to_lds(g0, g1, zz, zz, z8, 0);
}

// ---------------------------------------------------------------------------
// Layer-1 prep: x (B,9,N) f32 -> XT1 (B,N,32) f16 (zero pad) + xx (B,N)
// ---------------------------------------------------------------------------
__global__ void prep0_kernel(const float* __restrict__ x,
                             _Float16* __restrict__ XT, float* __restrict__ xx,
                             int N, int total) {
  int t = blockIdx.x * blockDim.x + threadIdx.x;
  if (t >= total) return;
  int b = t / N, n = t % N;
  float s = 0.f;
  for (int c = 0; c < 32; ++c) {
    float v = (c < 9) ? x[((size_t)b * 9 + c) * N + n] : 0.f;
    s += v * v;
    XT[((size_t)b * N + n) * 32 + c] = (_Float16)v;
  }
  xx[t] = s;
}

// ||x||^2 over a C-channel slice of a point-major f16 tensor (pitch elems/row)
__global__ void xx_f16_kernel(const _Float16* __restrict__ XT, int pitch, int C,
                              float* __restrict__ xx, int N, int total) {
  int t = blockIdx.x * blockDim.x + threadIdx.x;
  if (t >= total) return;
  int b = t / N, n = t % N;
  const _Float16* row = XT + ((size_t)b * N + n) * (size_t)pitch;
  float s = 0.f;
  for (int c = 0; c < C; ++c) { float v = (float)row[c]; s += v * v; }
  xx[t] = s;
}

// ---------------------------------------------------------------------------
// Fused Gram (WMMA) + top-k.  grid (N/16, B), block 32 (one wave).
// A (row tile) fragments live in registers for the whole sweep; column tiles
// stream through LDS via double-buffered TDM loads (TENSORcnt pipelining).
// pdist[m][n] = 2*<x_m,x_n> - xx[m] - xx[n]; never materializes N^2.
// ---------------------------------------------------------------------------
template <int CP>
__global__ void knn_topk_kernel(const _Float16* __restrict__ XT, int pitch,
                                const float* __restrict__ xx,
                                int* __restrict__ idx_out, int N) {
  constexpr int CHN = CP / 32;
  const int b    = blockIdx.y;
  const int m0   = blockIdx.x * 16;
  const int lane = threadIdx.x;
  const int half = lane >> 4;
  const int l16  = lane & 15;

  __shared__ _Float16 stage[2][16 * CP];
  __shared__ float tile[16][17];
  __shared__ float bestv[16][NK];
  __shared__ int   besti[16][NK];

  const _Float16* XTb = XT + (size_t)b * N * (size_t)pitch;
  const float*    xxb = xx + (size_t)b * N;
  const unsigned  lds0 = (unsigned)(uintptr_t)(&stage[0][0]);
  const unsigned  lds1 = (unsigned)(uintptr_t)(&stage[1][0]);

  // kick off column tile 0 while we set up
  tdm_load_tile_f16(lds0, XTb, CP, 16, (unsigned long long)pitch);

  if (lane < 16)
    for (int t = 0; t < NK; ++t) { bestv[lane][t] = -3.0e38f; besti[lane][t] = 0; }

  v16h afr[CHN];
#pragma unroll
  for (int c = 0; c < CHN; ++c)
#pragma unroll
    for (int e = 0; e < 16; ++e) {
      int k = c * 32 + (half << 4) + e;
      afr[c][e] = XTb[(size_t)(m0 + l16) * pitch + k];
    }
  float xxm[8];
#pragma unroll
  for (int i = 0; i < 8; ++i) xxm[i] = xxb[m0 + i + (half << 3)];

  const int TILES = N / 16;
  for (int t = 0; t < TILES; ++t) {
    if (t + 1 < TILES) {
      tdm_load_tile_f16((t & 1) ? lds0 : lds1,
                        XTb + (size_t)(t + 1) * 16 * pitch, CP, 16,
                        (unsigned long long)pitch);
      __builtin_amdgcn_s_wait_tensorcnt(1);  // tile t resident, t+1 in flight
    } else {
      __builtin_amdgcn_s_wait_tensorcnt(0);
    }
    const _Float16* S = &stage[t & 1][0];
    const int n0 = t * 16;
    v8f acc = {};
#pragma unroll
    for (int c = 0; c < CHN; ++c) {
      v16h bf;
#pragma unroll
      for (int e = 0; e < 16; ++e)
        bf[e] = S[l16 * CP + c * 32 + (half << 4) + e];
      acc = __builtin_amdgcn_wmma_f32_16x16x32_f16(false, afr[c], false, bf,
                                                   (short)0, acc, false, false);
    }
    const float xxn = xxb[n0 + l16];
#pragma unroll
    for (int i = 0; i < 8; ++i)
      tile[i + (half << 3)][l16] = 2.0f * acc[i] - xxm[i] - xxn;
    __syncthreads();
    if (lane < 16) {
      for (int nn = 0; nn < 16; ++nn) {
        float v = tile[lane][nn];
        if (v > bestv[lane][NK - 1]) {
          int q = NK - 1;
          while (q > 0 && bestv[lane][q - 1] < v) {
            bestv[lane][q] = bestv[lane][q - 1];
            besti[lane][q] = besti[lane][q - 1];
            --q;
          }
          bestv[lane][q] = v;
          besti[lane][q] = n0 + nn;
        }
      }
    }
    __syncthreads();
  }
  if (lane < 16)
    for (int t = 0; t < NK; ++t)
      idx_out[((size_t)b * N + m0 + lane) * NK + t] = besti[lane][t];
}

// ---------------------------------------------------------------------------
// Fused edge conv (templated on input channels so weight fragments are hoisted
// into registers once). Writes f16 channel slice of point-major concat buffer.
// ---------------------------------------------------------------------------
template <int CIN>
__global__ void edge_conv_kernel(const _Float16* __restrict__ XTin, int pitch,
                                 const int* __restrict__ knn,
                                 const float* __restrict__ W,
                                 const float* __restrict__ g, const float* __restrict__ bb,
                                 const float* __restrict__ mm, const float* __restrict__ vv,
                                 _Float16* __restrict__ out_t, int opitch, int c0out,
                                 int N) {
  constexpr int KF  = 2 * CIN;
  constexpr int CHN = (KF + 31) / 32;
  const int b    = blockIdx.z;
  const int m0   = blockIdx.y * 16;
  const int n0   = blockIdx.x * 16;
  const int lane = threadIdx.x;
  const int half = lane >> 4;
  const int l16  = lane & 15;
  const int n    = n0 + l16;
  const _Float16* XTb = XTin + (size_t)b * N * (size_t)pitch;

  v16h aw[CHN];
#pragma unroll
  for (int c = 0; c < CHN; ++c)
#pragma unroll
    for (int e = 0; e < 16; ++e) {
      int k = c * 32 + (half << 4) + e;
      aw[c][e] = (k < KF) ? (_Float16)W[(size_t)(m0 + l16) * KF + k] : (_Float16)0.f;
    }
  float sc[8], mu[8], bo[8];
#pragma unroll
  for (int i = 0; i < 8; ++i) {
    int m = m0 + i + (half << 3);
    sc[i] = g[m] * rsqrtf(vv[m] + BN_EPS);
    mu[i] = mm[m];
    bo[i] = bb[m];
  }
  int nbs[NK];
#pragma unroll
  for (int j = 0; j < NK; ++j) nbs[j] = knn[((size_t)b * N + n) * NK + j];
  __builtin_prefetch(XTb + (size_t)nbs[0] * pitch, 0, 1);

  v8f vmax;
#pragma unroll
  for (int i = 0; i < 8; ++i) vmax[i] = -3.0e38f;

  for (int j = 0; j < NK; ++j) {
    if (j + 1 < NK) __builtin_prefetch(XTb + (size_t)nbs[j + 1] * pitch, 0, 1);
    const int nb = nbs[j];
    v8f acc = {};
#pragma unroll
    for (int c = 0; c < CHN; ++c) {
      v16h bf;
#pragma unroll
      for (int e = 0; e < 16; ++e) {
        int k = c * 32 + (half << 4) + e;
        float val = 0.f;
        if (k < CIN)
          val = (float)XTb[(size_t)nb * pitch + k] - (float)XTb[(size_t)n * pitch + k];
        else if (k < KF)
          val = (float)XTb[(size_t)n * pitch + (k - CIN)];
        bf[e] = (_Float16)val;
      }
      acc = __builtin_amdgcn_wmma_f32_16x16x32_f16(false, aw[c], false, bf,
                                                   (short)0, acc, false, false);
    }
#pragma unroll
    for (int i = 0; i < 8; ++i) {
      float y = (acc[i] - mu[i]) * sc[i] + bo[i];
      y = (y >= 0.f) ? y : SLOPE * y;
      vmax[i] = fmaxf(vmax[i], y);
    }
  }
#pragma unroll
  for (int i = 0; i < 8; ++i) {
    int m = m0 + i + (half << 3);
    out_t[((size_t)b * N + n) * opitch + c0out + m] = (_Float16)vmax[i];
  }
}

// ---------------------------------------------------------------------------
// Dense 1x1 conv, C=512: Yt(B,N,O) f16 = lrelu(bn(W @ Xt^T [+bias])).
// Point-major f16 input -> contiguous B-fragment loads.
// ---------------------------------------------------------------------------
__global__ void gemm512_t_kernel(const _Float16* __restrict__ Xt,
                                 const float* __restrict__ W,
                                 const float* __restrict__ bias,
                                 const float* __restrict__ g, const float* __restrict__ bb,
                                 const float* __restrict__ mm, const float* __restrict__ vv,
                                 _Float16* __restrict__ Yt, int N, int O) {
  constexpr int C = 512;
  const int b    = blockIdx.z;
  const int m0   = blockIdx.y * 16;
  const int n0   = blockIdx.x * 16;
  const int lane = threadIdx.x;
  const int half = lane >> 4;
  const int l16  = lane & 15;
  const int n    = n0 + l16;
  const _Float16* xrow = Xt + ((size_t)b * N + n) * C;

  v8f acc = {};
#pragma unroll
  for (int c = 0; c < C / 32; ++c) {
    v16h a, bf;
#pragma unroll
    for (int e = 0; e < 16; ++e) {
      int k = c * 32 + (half << 4) + e;
      a[e] = (_Float16)W[(size_t)(m0 + l16) * C + k];
    }
#pragma unroll
    for (int e = 0; e < 16; ++e)
      bf[e] = xrow[c * 32 + (half << 4) + e];
    acc = __builtin_amdgcn_wmma_f32_16x16x32_f16(false, a, false, bf,
                                                 (short)0, acc, false, false);
  }
#pragma unroll
  for (int i = 0; i < 8; ++i) {
    int m = m0 + i + (half << 3);
    float y = acc[i] + (bias ? bias[m] : 0.f);
    y = (y - mm[m]) * (g[m] * rsqrtf(vv[m] + BN_EPS)) + bb[m];
    y = (y >= 0.f) ? y : SLOPE * y;
    Yt[((size_t)b * N + n) * O + m] = (_Float16)y;
  }
}

// Classifier: out(B,N,20) f32 = Xt(B,N,512) @ sem_w^T + sem_b (WMMA, M-guard).
__global__ void sem_kernel(const _Float16* __restrict__ Xt, const float* __restrict__ sw,
                           const float* __restrict__ sb, float* __restrict__ out,
                           int N, int O) {
  constexpr int C = 512;
  const int b    = blockIdx.z;
  const int m0   = blockIdx.y * 16;
  const int n0   = blockIdx.x * 16;
  const int lane = threadIdx.x;
  const int half = lane >> 4;
  const int l16  = lane & 15;
  const int n    = n0 + l16;
  const int orow = m0 + l16;
  const _Float16* xrow = Xt + ((size_t)b * N + n) * C;

  v8f acc = {};
#pragma unroll
  for (int c = 0; c < C / 32; ++c) {
    v16h a, bf;
#pragma unroll
    for (int e = 0; e < 16; ++e) {
      int k = c * 32 + (half << 4) + e;
      a[e] = (orow < O) ? (_Float16)sw[(size_t)orow * C + k] : (_Float16)0.f;
    }
#pragma unroll
    for (int e = 0; e < 16; ++e)
      bf[e] = xrow[c * 32 + (half << 4) + e];
    acc = __builtin_amdgcn_wmma_f32_16x16x32_f16(false, a, false, bf,
                                                 (short)0, acc, false, false);
  }
#pragma unroll
  for (int i = 0; i < 8; ++i) {
    int m = m0 + i + (half << 3);
    if (m < O) out[((size_t)b * N + n) * O + m] = acc[i] + sb[m];
  }
}

// ---------------------------------------------------------------------------
// Input flattening assumption: jax pytree order (sorted dict keys, recursive).
// ---------------------------------------------------------------------------
enum {
  IN_C5_B = 0, IN_C5_G, IN_C5_M, IN_C5_V, IN_C5_W,
  IN_EC1_B, IN_EC1_G, IN_EC1_M, IN_EC1_V, IN_EC1_W,
  IN_EC2_B, IN_EC2_G, IN_EC2_M, IN_EC2_V, IN_EC2_W,
  IN_EC3_B, IN_EC3_G, IN_EC3_M, IN_EC3_V, IN_EC3_W,
  IN_EC4_B, IN_EC4_G, IN_EC4_M, IN_EC4_V, IN_EC4_W,
  IN_H0_B, IN_H0_BIAS, IN_H0_G, IN_H0_M, IN_H0_V, IN_H0_W,
  IN_H1_B, IN_H1_BIAS, IN_H1_G, IN_H1_M, IN_H1_V, IN_H1_W,
  IN_H2_B, IN_H2_BIAS, IN_H2_G, IN_H2_M, IN_H2_V, IN_H2_W,
  IN_H3_B, IN_H3_BIAS, IN_H3_G, IN_H3_M, IN_H3_V, IN_H3_W,
  IN_SEM_B, IN_SEM_W,
  IN_X
};

extern "C" void kernel_launch(void* const* d_in, const int* in_sizes, int n_in,
                              void* d_out, int out_size, void* d_ws, size_t ws_size,
                              hipStream_t stream) {
  const int B = 4, N = 4096, CTOT = 512;
  (void)in_sizes; (void)n_in; (void)out_size; (void)ws_size;

  auto F = [&](int i) { return (const float*)d_in[i]; };

  char* wsb = (char*)d_ws;
  size_t off = 0;
  auto carve = [&](size_t bytes) -> void* {
    void* p = wsb + off;
    off = (off + bytes + 255) & ~(size_t)255;
    return p;
  };
  float*    xx     = (float*)    carve((size_t)B * N * sizeof(float));
  _Float16* XT1    = (_Float16*) carve((size_t)B * N * 32 * sizeof(_Float16));
  int*      kidx   = (int*)      carve((size_t)B * N * NK * sizeof(int));
  _Float16* cat_t  = (_Float16*) carve((size_t)B * N * CTOT * sizeof(_Float16));
  _Float16* x5_t   = (_Float16*) carve((size_t)B * N * CTOT * sizeof(_Float16));
  _Float16* feat_t = (_Float16*) carve((size_t)B * N * CTOT * sizeof(_Float16));

  const int  total = B * N;
  const dim3 blk32(32, 1, 1);
  const dim3 gknn(N / 16, B, 1);
  const int  eb = (total + 255) / 256;

  // ---- layer 1 (C=9, pad 32): prep from x, knn, edge -> cat_t[:, 0:64) ----
  prep0_kernel<<<eb, 256, 0, stream>>>(F(IN_X), XT1, xx, N, total);
  knn_topk_kernel<32><<<gknn, blk32, 0, stream>>>(XT1, 32, xx, kidx, N);
  edge_conv_kernel<9><<<dim3(N / 16, 4, B), blk32, 0, stream>>>(
      XT1, 32, kidx, F(IN_EC1_W), F(IN_EC1_G), F(IN_EC1_B), F(IN_EC1_M), F(IN_EC1_V),
      cat_t, CTOT, 0, N);

  // ---- layer 2 (C=64 from cat_t[:,0:64)) -> cat_t[:, 64:128) ----
  xx_f16_kernel<<<eb, 256, 0, stream>>>(cat_t + 0, CTOT, 64, xx, N, total);
  knn_topk_kernel<64><<<gknn, blk32, 0, stream>>>(cat_t + 0, CTOT, xx, kidx, N);
  edge_conv_kernel<64><<<dim3(N / 16, 4, B), blk32, 0, stream>>>(
      cat_t + 0, CTOT, kidx, F(IN_EC2_W), F(IN_EC2_G), F(IN_EC2_B), F(IN_EC2_M), F(IN_EC2_V),
      cat_t, CTOT, 64, N);

  // ---- layer 3 (C=64 from cat_t[:,64:128)) -> cat_t[:, 128:256) ----
  xx_f16_kernel<<<eb, 256, 0, stream>>>(cat_t + 64, CTOT, 64, xx, N, total);
  knn_topk_kernel<64><<<gknn, blk32, 0, stream>>>(cat_t + 64, CTOT, xx, kidx, N);
  edge_conv_kernel<64><<<dim3(N / 16, 8, B), blk32, 0, stream>>>(
      cat_t + 64, CTOT, kidx, F(IN_EC3_W), F(IN_EC3_G), F(IN_EC3_B), F(IN_EC3_M), F(IN_EC3_V),
      cat_t, CTOT, 128, N);

  // ---- layer 4 (C=128 from cat_t[:,128:256)) -> cat_t[:, 256:512) ----
  xx_f16_kernel<<<eb, 256, 0, stream>>>(cat_t + 128, CTOT, 128, xx, N, total);
  knn_topk_kernel<128><<<gknn, blk32, 0, stream>>>(cat_t + 128, CTOT, xx, kidx, N);
  edge_conv_kernel<128><<<dim3(N / 16, 16, B), blk32, 0, stream>>>(
      cat_t + 128, CTOT, kidx, F(IN_EC4_W), F(IN_EC4_G), F(IN_EC4_B), F(IN_EC4_M), F(IN_EC4_V),
      cat_t, CTOT, 256, N);

  // ---- c5: x5 = lrelu(bn(W5 @ cat)) ----
  gemm512_t_kernel<<<dim3(N / 16, 32, B), blk32, 0, stream>>>(
      cat_t, F(IN_C5_W), nullptr,
      F(IN_C5_G), F(IN_C5_B), F(IN_C5_M), F(IN_C5_V), x5_t, N, 512);

  // ---- heads: feat overwritten each iter in reference -> only head 3 matters ----
  gemm512_t_kernel<<<dim3(N / 16, 32, B), blk32, 0, stream>>>(
      x5_t, F(IN_H3_W), F(IN_H3_BIAS),
      F(IN_H3_G), F(IN_H3_B), F(IN_H3_M), F(IN_H3_V), feat_t, N, 512);

  // ---- sem: out (B,N,20) f32 ----
  sem_kernel<<<dim3(N / 16, 2, B), blk32, 0, stream>>>(
      feat_t, F(IN_SEM_W), F(IN_SEM_B), (float*)d_out, N, 20);
}
